// Attention_14594298871901
// MI455X (gfx1250) — matrix operand order repaired and checked by reading, not compiled, not run on hardware
//
#include <hip/hip_runtime.h>

// ---------------------------------------------------------------------------
// Types for CDNA5 WMMA (gfx1250, wave32)
// ---------------------------------------------------------------------------
typedef __bf16 v16bf __attribute__((ext_vector_type(16)));
typedef __bf16 v8bf  __attribute__((ext_vector_type(8)));
typedef float  v8f   __attribute__((ext_vector_type(8)));

#define H_    12
#define HD_   64
#define NSEQ  1024
#define BB    8
#define CDIM  768
#define LDSTRIDE 40   // 32 K-elems + 8 pad (80B rows, 16B aligned)

// ---------------------------------------------------------------------------
// CDNA5 async global->LDS copy (ASYNCcnt-tracked), 16 bytes per lane.
// VDST VGPR carries the wave-relative LDS byte offset (low 32 bits of the
// flat shared-memory address, ISA 10.2 aperture mapping).
// ---------------------------------------------------------------------------
__device__ __forceinline__ void async_cp16o(unsigned lds_off, const void* g) {
  asm volatile("global_load_async_to_lds_b128 %0, %1, off"
               :: "v"(lds_off), "v"((unsigned long long)g) : "memory");
}
__device__ __forceinline__ void async_wait0() {
  asm volatile("s_wait_asynccnt 0x0" ::: "memory");
}
__device__ __forceinline__ unsigned lds_off32(const void* p) {
  return (unsigned)(unsigned long long)p;
}

// ---------------------------------------------------------------------------
// fp32 -> bf16 conversion (grid-stride)
// ---------------------------------------------------------------------------
__global__ void cvt_f32_to_bf16(const float* __restrict__ src,
                                __bf16* __restrict__ dst, int n) {
  int i = blockIdx.x * blockDim.x + threadIdx.x;
  int stride = gridDim.x * blockDim.x;
  for (; i < n; i += stride) dst[i] = (__bf16)src[i];
}

// ---------------------------------------------------------------------------
// In-place row softmax over 1024-wide rows: one wave32 per row, shuffle
// reductions only (no LDS, no barriers).
// ---------------------------------------------------------------------------
__global__ void softmax_rows_1024(float* __restrict__ attn, int rows) {
  const int wid  = threadIdx.x >> 5;
  const int lane = threadIdx.x & 31;
  const int row  = blockIdx.x * 8 + wid;
  if (row >= rows) return;
  float* p = attn + (size_t)row * 1024 + lane * 4;
  float4 v[8];
  float m = -3.402823466e38f;
#pragma unroll
  for (int i = 0; i < 8; ++i) {
    v[i] = *(const float4*)(p + i * 128);
    m = fmaxf(m, fmaxf(fmaxf(v[i].x, v[i].y), fmaxf(v[i].z, v[i].w)));
  }
#pragma unroll
  for (int off = 16; off > 0; off >>= 1) m = fmaxf(m, __shfl_xor(m, off, 32));
  float s = 0.f;
#pragma unroll
  for (int i = 0; i < 8; ++i) {
    v[i].x = __expf(v[i].x - m); v[i].y = __expf(v[i].y - m);
    v[i].z = __expf(v[i].z - m); v[i].w = __expf(v[i].w - m);
    s += v[i].x + v[i].y + v[i].z + v[i].w;
  }
#pragma unroll
  for (int off = 16; off > 0; off >>= 1) s += __shfl_xor(s, off, 32);
  const float inv = 1.f / s;
#pragma unroll
  for (int i = 0; i < 8; ++i) {
    v[i].x *= inv; v[i].y *= inv; v[i].z *= inv; v[i].w *= inv;
    *(float4*)(p + i * 128) = v[i];
  }
}

// ---------------------------------------------------------------------------
// Generic batched GEMM: C[z] = A[z] (MxK) * W[z] (NpxK)^T with WMMA bf16.
// ASYNCP=true : bf16 A, exact 128-multiple M/Np, K multiple of 64;
//               double-buffered LDS fed by global_load_async_to_lds_b128,
//               branchless steady-state loop with peeled tail.
// ASYNCP=false: A may be fp32 (converted while staging), Np bounded.
// ---------------------------------------------------------------------------
struct GemmP {
  const void*   A;
  const __bf16* W;
  long long strideA, strideW, strideC;   // per-z element strides
  int lda, ldw, ldc, M, Np, K;
  float* outF;
  const float* bias;
  __bf16 *qp, *kp, *vtp, *op;
  float scale;
};

enum { EPI_F32 = 0, EPI_QKV = 1, EPI_OHEADS = 2, EPI_BIAS = 3 };

__device__ __forceinline__ v8bf load8(const __bf16* p) {
  return *(const v8bf*)p;
}
__device__ __forceinline__ v8bf load8(const float* p) {
  const float4 a = *(const float4*)p;
  const float4 b = *(const float4*)(p + 4);
  v8bf o;
  o[0] = (__bf16)a.x; o[1] = (__bf16)a.y; o[2] = (__bf16)a.z; o[3] = (__bf16)a.w;
  o[4] = (__bf16)b.x; o[5] = (__bf16)b.y; o[6] = (__bf16)b.z; o[7] = (__bf16)b.w;
  return o;
}
__device__ __forceinline__ v8bf zero8() {
  v8bf o;
#pragma unroll
  for (int e = 0; e < 8; ++e) o[e] = (__bf16)0.0f;
  return o;
}

// A fragment 16x32: lane half 0 holds K[0..7]+K[16..23], half 1 K[8..15]+K[24..31]
__device__ __forceinline__ v16bf fragA(const __bf16* lds, int rowbase, int lane) {
  const int half = lane >> 4, r = lane & 15;
  const __bf16* p = lds + (rowbase + r) * LDSTRIDE + half * 8;
  v8bf lo = *(const v8bf*)p;
  v8bf hi = *(const v8bf*)(p + 16);
  return __builtin_shufflevector(lo, hi, 0,1,2,3,4,5,6,7,8,9,10,11,12,13,14,15);
}
// B fragment 32x16 (as rows of W): lane half 0 holds K[0..7]+K[16..23], half 1 K[8..15]+K[24..31]
__device__ __forceinline__ v16bf fragB(const __bf16* lds, int rowbase, int lane) {
  const int half = lane >> 4, r = lane & 15;
  const __bf16* p = lds + (rowbase + r) * LDSTRIDE + half * 8;
  v8bf lo = *(const v8bf*)p;
  v8bf hi = *(const v8bf*)(p + 16);
  return __builtin_shufflevector(lo, hi, 0,1,2,3,4,5,6,7,8,9,10,11,12,13,14,15);
}

template <typename TA, int EPI, bool ASYNCP>
__global__ void __launch_bounds__(256) k_gemm(GemmP p) {
  __shared__ __bf16 ldsA[2][128 * LDSTRIDE];
  __shared__ __bf16 ldsB[2][128 * LDSTRIDE];

  const int t    = threadIdx.x;
  const int lane = t & 31;
  const int wid  = t >> 5;
  const int wm   = wid >> 1;            // 0..3: 32-row strip
  const int wn   = wid & 1;             // 0..1: 64-col strip
  const int m0   = blockIdx.y * 128;
  const int n0   = blockIdx.x * 128;
  const int z    = blockIdx.z;

  const TA*     Ag = (const TA*)p.A + (size_t)z * p.strideA;
  const __bf16* Wg = p.W + (size_t)z * p.strideW;

  v8f acc[2][4];
#pragma unroll
  for (int i = 0; i < 2; ++i)
#pragma unroll
    for (int j = 0; j < 4; ++j)
#pragma unroll
      for (int v = 0; v < 8; ++v) acc[i][j][v] = 0.f;

  auto compute_tile = [&](int buf) {
    v16bf af[2], bfr[4];
#pragma unroll
    for (int i = 0; i < 2; ++i) af[i] = fragA(ldsA[buf], wm * 32 + i * 16, lane);
#pragma unroll
    for (int j = 0; j < 4; ++j) bfr[j] = fragB(ldsB[buf], wn * 64 + j * 16, lane);
#pragma unroll
    for (int i = 0; i < 2; ++i)
#pragma unroll
      for (int j = 0; j < 4; ++j)
        acc[i][j] = __builtin_amdgcn_wmma_f32_16x16x32_bf16(
            false, af[i], false, bfr[j], (short)0, acc[i][j], false, false);
  };

  if (ASYNCP) {
    // ---- async double-buffered pipeline; K is a multiple of 64 ----
    // Per-thread staging slots: rows (t>>2) and (t>>2)+64, columns (t&3)*8.
    const int srow = t >> 2, scol = (t & 3) << 3;
    const TA* gA0 = Ag + (size_t)(m0 + srow) * p.lda + scol;
    const TA* gA1 = Ag + (size_t)(m0 + srow + 64) * p.lda + scol;
    const __bf16* gB0 = Wg + (size_t)(n0 + srow) * p.ldw + scol;
    const __bf16* gB1 = Wg + (size_t)(n0 + srow + 64) * p.ldw + scol;
    unsigned lA0[2], lA1[2], lB0[2], lB1[2];
#pragma unroll
    for (int b = 0; b < 2; ++b) {
      lA0[b] = lds_off32(&ldsA[b][srow * LDSTRIDE + scol]);
      lA1[b] = lds_off32(&ldsA[b][(srow + 64) * LDSTRIDE + scol]);
      lB0[b] = lds_off32(&ldsB[b][srow * LDSTRIDE + scol]);
      lB1[b] = lds_off32(&ldsB[b][(srow + 64) * LDSTRIDE + scol]);
    }
    auto stage = [&](int b) {   // stages next 32-deep tile, advances pointers
      async_cp16o(lA0[b], gA0);
      async_cp16o(lA1[b], gA1);
      async_cp16o(lB0[b], gB0);
      async_cp16o(lB1[b], gB1);
      gA0 += 32; gA1 += 32; gB0 += 32; gB1 += 32;
    };

    stage(0);                                   // tile 0 -> buf0
    for (int k0 = 0; k0 + 64 < p.K; k0 += 64) { // branchless steady state
      async_wait0();                            // own tile copies complete
      __syncthreads();                          // tile visible; prev reads done
      stage(1);                                 // flies under the WMMAs
      compute_tile(0);

      async_wait0();
      __syncthreads();
      stage(0);
      compute_tile(1);
    }
    // tail: last two tiles
    async_wait0();
    __syncthreads();
    stage(1);
    compute_tile(0);

    async_wait0();
    __syncthreads();
    compute_tile(1);
  } else {
    // ---- synchronous path (fp32 A conversion, bounded Np) ----
    for (int k0 = 0; k0 < p.K; k0 += 32) {
      __syncthreads();
      v8bf ra[2], rw[2];
#pragma unroll
      for (int u = 0; u < 2; ++u) {
        const int s = t + u * 256;
        const int row = s >> 2, col = (s & 3) << 3;
        ra[u] = load8(Ag + (size_t)(m0 + row) * p.lda + k0 + col);
        rw[u] = (n0 + row < p.Np)
                    ? load8(Wg + (size_t)(n0 + row) * p.ldw + k0 + col)
                    : zero8();
      }
#pragma unroll
      for (int u = 0; u < 2; ++u) {
        const int s = t + u * 256;
        const int row = s >> 2, col = (s & 3) << 3;
        *(v8bf*)&ldsA[0][row * LDSTRIDE + col] = ra[u];
        *(v8bf*)&ldsB[0][row * LDSTRIDE + col] = rw[u];
      }
      __syncthreads();
      compute_tile(0);
    }
  }

  // ---- epilogue: C/D layout: VGPR v, lanes 0-15 -> M=v, lanes 16-31 -> M=v+8
  const int colL = lane & 15;
  const int rowH = (lane >> 4) * 8;
#pragma unroll
  for (int i = 0; i < 2; ++i) {
#pragma unroll
    for (int j = 0; j < 4; ++j) {
      const int gc = n0 + wn * 64 + j * 16 + colL;
#pragma unroll
      for (int v = 0; v < 8; ++v) {
        const int gr = m0 + wm * 32 + i * 16 + rowH + v;
        if (!ASYNCP && (gr >= p.M || gc >= p.Np)) continue;
        const float val = acc[i][j][v];
        if (EPI == EPI_F32) {
          // streaming 400MB score write: non-temporal
          __builtin_nontemporal_store(
              val, p.outF + (size_t)z * p.strideC + (size_t)gr * p.ldc + gc);
        } else if (EPI == EPI_BIAS) {
          p.outF[(size_t)gr * p.ldc + gc] = val + p.bias[gc];
        } else if (EPI == EPI_QKV) {
          const int which = gc / CDIM;
          const int cc = gc - which * CDIM;
          const int h = cc >> 6, d = cc & 63;
          const int b = gr >> 10, n = gr & 1023;
          const size_t bh = (size_t)(b * H_ + h);
          if (which == 0)
            p.qp[(bh * NSEQ + n) * HD_ + d] = (__bf16)(val * p.scale);
          else if (which == 1)
            p.kp[(bh * NSEQ + n) * HD_ + d] = (__bf16)val;
          else
            p.vtp[(bh * HD_ + d) * NSEQ + n] = (__bf16)val;   // V transposed
        } else { // EPI_OHEADS: O in [B,N,C] layout, bf16
          const int b = z / H_;
          const int h = z - b * H_;
          p.op[((size_t)b * NSEQ + gr) * CDIM + h * HD_ + gc] = (__bf16)val;
        }
      }
    }
  }
}

// ---------------------------------------------------------------------------
// Launch
// ---------------------------------------------------------------------------
extern "C" void kernel_launch(void* const* d_in, const int* in_sizes, int n_in,
                              void* d_out, int out_size, void* d_ws, size_t ws_size,
                              hipStream_t stream) {
  (void)in_sizes; (void)n_in; (void)out_size; (void)ws_size;
  const float* x      = (const float*)d_in[0];   // [8,1024,768]
  const float* w_qkv  = (const float*)d_in[1];   // [2304,768]
  const float* w_proj = (const float*)d_in[2];   // [768,768]
  const float* b_proj = (const float*)d_in[3];   // [768]

  float* outp = (float*)d_out;                           // [8,1024,768]
  float* attn = outp + (size_t)BB * NSEQ * CDIM;         // [8,12,1024,1024]

  auto align = [](size_t v) { return (v + 255) & ~(size_t)255; };
  const size_t tokB = (size_t)BB * NSEQ * CDIM * sizeof(__bf16);
  char* w = (char*)d_ws;
  __bf16* xb     = (__bf16*)w; w += align(tokB);
  __bf16* wqkvb  = (__bf16*)w; w += align((size_t)3 * CDIM * CDIM * 2);
  __bf16* wprojb = (__bf16*)w; w += align((size_t)CDIM * CDIM * 2);
  __bf16* Qb     = (__bf16*)w; w += align(tokB);   // [B,H,N,HD] pre-scaled
  __bf16* Kb     = (__bf16*)w; w += align(tokB);   // [B,H,N,HD]
  __bf16* Vtb    = (__bf16*)w; w += align(tokB);   // [B,H,HD,N]
  __bf16* Ob     = (__bf16*)w; w += align(tokB);   // [B,N,C]

  cvt_f32_to_bf16<<<4096, 256, 0, stream>>>(x,      xb,     BB * NSEQ * CDIM);
  cvt_f32_to_bf16<<<2048, 256, 0, stream>>>(w_qkv,  wqkvb,  3 * CDIM * CDIM);
  cvt_f32_to_bf16<<<1024, 256, 0, stream>>>(w_proj, wprojb, CDIM * CDIM);

  { // QKV = X @ Wqkv^T, scattered to Q(scaled)/K/V^T
    GemmP p = {}; p.A = xb; p.W = wqkvb; p.lda = CDIM; p.ldw = CDIM;
    p.M = BB * NSEQ; p.Np = 3 * CDIM; p.K = CDIM;
    p.qp = Qb; p.kp = Kb; p.vtp = Vtb; p.scale = 0.125f;  // HD^-0.5
    k_gemm<__bf16, EPI_QKV, true><<<dim3(18, 64, 1), 256, 0, stream>>>(p);
  }
  { // scores = Qs @ K^T per (b,h)  ->  attn region (fp32 logits)
    GemmP p = {}; p.A = Qb; p.W = Kb; p.lda = HD_; p.ldw = HD_;
    p.strideA = NSEQ * HD_; p.strideW = NSEQ * HD_;
    p.M = NSEQ; p.Np = NSEQ; p.K = HD_;
    p.outF = attn; p.strideC = (long long)NSEQ * NSEQ; p.ldc = NSEQ;
    k_gemm<__bf16, EPI_F32, true><<<dim3(8, 8, BB * H_), 256, 0, stream>>>(p);
  }
  softmax_rows_1024<<<(BB * H_ * NSEQ) / 8, 256, 0, stream>>>(attn, BB * H_ * NSEQ);
  { // O = attn @ V per (b,h); attn is fp32, converted to bf16 while staging
    GemmP p = {}; p.A = attn; p.W = Vtb; p.lda = NSEQ; p.ldw = NSEQ;
    p.strideA = (long long)NSEQ * NSEQ; p.strideW = HD_ * NSEQ;
    p.M = NSEQ; p.Np = HD_; p.K = NSEQ;
    p.op = Ob;
    k_gemm<float, EPI_OHEADS, false><<<dim3(1, 8, BB * H_), 256, 0, stream>>>(p);
  }
  { // out = O @ Wproj^T + b
    GemmP p = {}; p.A = Ob; p.W = wprojb; p.lda = CDIM; p.ldw = CDIM;
    p.M = BB * NSEQ; p.Np = CDIM; p.K = CDIM;
    p.outF = outp; p.ldc = CDIM; p.bias = b_proj;
    k_gemm<__bf16, EPI_BIAS, true><<<dim3(6, 64, 1), 256, 0, stream>>>(p);
  }
}
